// MultiHeadAttention_75857712382707
// MI455X (gfx1250) — compile-verified, matrix operand order checked by reading, not run on hardware
//
#include <hip/hip_runtime.h>
#include <hip/hip_bf16.h>
#include <cstdint>

#define B_ 4
#define T_ 2048
#define C_ 384
#define H_ 6
#define D_ 64

typedef __attribute__((ext_vector_type(16))) _Float16 v16h;
typedef __attribute__((ext_vector_type(8)))  _Float16 v8h;
typedef __attribute__((ext_vector_type(8)))  float    v8f;

union Frag { v16h v; v8h h[2]; };

// ---------------------------------------------------------------------------
// CDNA5 async global->LDS staging (ASYNCcnt path)
// ---------------------------------------------------------------------------
__device__ __forceinline__ void async_ld_b128(uint32_t lds_byte_off, const void* gptr) {
  asm volatile("global_load_async_to_lds_b128 %0, %1, off"
               :: "v"(lds_byte_off), "v"(gptr) : "memory");
}
__device__ __forceinline__ void wait_async0() {
#if __has_builtin(__builtin_amdgcn_s_wait_asynccnt)
  __builtin_amdgcn_s_wait_asynccnt(0);
#else
  asm volatile("s_wait_asynccnt 0" ::: "memory");
#endif
}
__device__ __forceinline__ uint32_t lds_off(const void* p) {
  // generic shared-aperture address keeps the LDS offset in addr[31:0]
  return (uint32_t)(uintptr_t)p;
}

__device__ __forceinline__ float clamp_bias(float x) {
  return fminf(2.0f, fmaxf(-2.0f, x));
}

// ---------------------------------------------------------------------------
// f32 -> f16 conversion
// ---------------------------------------------------------------------------
__global__ void cvt_kernel(const float* __restrict__ src, _Float16* __restrict__ dst, int n) {
  int i = blockIdx.x * blockDim.x + threadIdx.x;
  int stride = gridDim.x * blockDim.x;
  for (; i < n; i += stride) dst[i] = (_Float16)src[i];
}

// ---------------------------------------------------------------------------
// QKV projection + RoPE + RMS-norm.
// grid: (T/16, 3*B); block: 128 (4 waves). Block computes all 384 output cols
// (all 6 heads) for 16 tokens of one matrix: 4 waves x 6 N-tiles, X loaded once.
// which = blockIdx.y / B : 0=Q, 1=K, 2=V
// Q/K stored [B,H,T,64] (Q pre-scaled by 1/sqrt(D)); V stored transposed [B,H,64,T].
// ---------------------------------------------------------------------------
__global__ __launch_bounds__(128)
void qkv_kernel(const _Float16* __restrict__ xh,
                const _Float16* __restrict__ wqh,
                const _Float16* __restrict__ wkh,
                const _Float16* __restrict__ wvh,
                const float* __restrict__ cosp,
                const float* __restrict__ sinp,
                _Float16* __restrict__ qh,
                _Float16* __restrict__ kh,
                _Float16* __restrict__ vh) {
  __shared__ __align__(16) _Float16 xs[16][392];    // 16 x 384 (+pad)  12.5 KB
  __shared__ __align__(16) float    outs[16][392];  // 16 x 384 (+pad)  25 KB

  const int tid   = threadIdx.x;
  const int b     = blockIdx.y % B_;
  const int which = blockIdx.y / B_;
  const int t0    = blockIdx.x * 16;

  const _Float16* xrow = xh + ((size_t)b * T_ + t0) * C_;
  for (int i = tid; i < 768; i += 128) {            // 768 x 16B chunks
    int r = i / 48, cc = (i % 48) * 8;
    async_ld_b128(lds_off(&xs[r][cc]), (const void*)(xrow + (size_t)r * C_ + cc));
  }
  wait_async0();
  __syncthreads();

  const _Float16* Wsel = (which == 0) ? wqh : (which == 1) ? wkh : wvh;
  const int wave  = tid >> 5;
  const int lane  = tid & 31;
  const int lhalf = lane & 15;
  const int hb    = (lane >> 4) * 8;   // K sub-block select per A/B layout
  const int mrb   = (lane >> 4) * 8;   // C/D row base

  for (int nt = 0; nt < 6; ++nt) {
    const int n0 = (nt * 4 + wave) * 16;            // output col base, 0..368
    const _Float16* wrow = Wsel + (size_t)(n0 + lhalf) * C_;
    v8f acc = {};
#pragma unroll
    for (int kk = 0; kk < 12; ++kk) {
      const int kb = kk * 32;
      Frag a, bf;
      a.h[0]  = *reinterpret_cast<const v8h*>(&xs[lhalf][kb + hb]);
      a.h[1]  = *reinterpret_cast<const v8h*>(&xs[lhalf][kb + 16 + hb]);
      bf.h[0] = *reinterpret_cast<const v8h*>(wrow + kb + hb);
      bf.h[1] = *reinterpret_cast<const v8h*>(wrow + kb + 16 + hb);
      acc = __builtin_amdgcn_wmma_f32_16x16x32_f16(false, a.v, false, bf.v,
                                                   (short)0, acc, false, false);
    }
#pragma unroll
    for (int r = 0; r < 8; ++r) outs[mrb + r][n0 + lhalf] = acc[r];
  }
  __syncthreads();

  if (which == 2) {
    // V transposed: [B,H,D,T]
    for (int i = tid; i < 16 * 384; i += 128) {
      int col = i >> 4, r = i & 15;
      int h = col >> 6, d = col & 63;
      vh[((size_t)(b * H_ + h) * D_ + d) * T_ + t0 + r] = (_Float16)outs[r][col];
    }
  } else {
    // RoPE + RMS: 384 work items = 6 heads x 16 rows x 4 parts; 3 uniform passes.
    _Float16* dstbase = (which == 0) ? qh : kh;
    const float qscale = (which == 0) ? 0.125f : 1.0f;  // fold softmax 1/sqrt(D) into Q
    for (int item = tid; item < 384; item += 128) {
      const int h    = item >> 6;
      const int rem  = item & 63;
      const int row  = rem >> 2;
      const int part = rem & 3;
      const int t    = t0 + row;
      float r1[8], r2[8];
      float ss = 0.0f;
#pragma unroll
      for (int jj = 0; jj < 8; ++jj) {
        const int j = part * 8 + jj;
        float c  = cosp[(size_t)t * 32 + j];
        float s  = sinp[(size_t)t * 32 + j];
        float x1 = outs[row][h * 64 + j];
        float x2 = outs[row][h * 64 + j + 32];
        r1[jj] = x1 * c + x2 * s;
        r2[jj] = -x1 * s + x2 * c;
        ss += r1[jj] * r1[jj] + r2[jj] * r2[jj];
      }
      // 4 parts of one row sit in 4 adjacent lanes (groups aligned to 4)
      ss += __shfl_xor(ss, 1, 32);
      ss += __shfl_xor(ss, 2, 32);
      const float inv = rsqrtf(ss * (1.0f / 64.0f) + 1.1920929e-7f) * qscale;
      _Float16* qout = dstbase + ((size_t)(b * H_ + h) * T_ + t) * D_;
#pragma unroll
      for (int jj = 0; jj < 8; ++jj) {
        const int j = part * 8 + jj;
        qout[j]      = (_Float16)(r1[jj] * inv);
        qout[j + 32] = (_Float16)(r2[jj] * inv);
      }
    }
  }
}

// ---------------------------------------------------------------------------
// Flash attention, fixed-max softmax.
// RMS-norm bounds: ||q_scaled|| = 1, ||k|| = 8 => |score| <= 8; bias clip +-2
// => |s| <= 10. exp(10)=2.2e4 < f16 max; row sum < 2048*e^10 ~ 4.5e7 < f32 max.
// So no running max, no accumulator rescale; row sum reduced once at the end.
// Q/K [B,H,T,64] f16 (Q pre-scaled), V [B,H,64,T] f16. Y f16 token-major [B,T,C].
// grid: (T/64, H, B); block: 128 (4 waves, 16 query rows each).
// ---------------------------------------------------------------------------
__global__ __launch_bounds__(128)
void attn_kernel(const _Float16* __restrict__ qh,
                 const _Float16* __restrict__ kh,
                 const _Float16* __restrict__ vh,
                 const int* __restrict__ maskp,
                 const float* __restrict__ rawb,
                 const int* __restrict__ bscale,
                 _Float16* __restrict__ yh) {
  __shared__ __align__(16) _Float16 kts[2][32][72];   // 32 keys x 64 dims (+pad)
  __shared__ __align__(16) _Float16 vtt[2][64][40];   // 64 dims x 32 keys (+pad)
  __shared__ __align__(16) _Float16 pts[4][16][40];   // per-wave P tile 16x32 (+pad)

  const int tid   = threadIdx.x;
  const int wave  = tid >> 5;
  const int lane  = tid & 31;
  const int lhalf = lane & 15;
  const int hb    = (lane >> 4) * 8;
  const int mrb   = (lane >> 4) * 8;

  const int b   = blockIdx.z;
  const int h   = blockIdx.y;
  const int qt0 = blockIdx.x * 64 + wave * 16;

  const _Float16* Qb  = qh + (size_t)(b * H_ + h) * T_ * D_;
  const _Float16* Kb  = kh + (size_t)(b * H_ + h) * T_ * D_;
  const _Float16* Vtb = vh + (size_t)(b * H_ + h) * D_ * T_;

  // Q fragments (K = 0..31 and 32..63)
  Frag qa0, qa1;
  {
    const _Float16* qr = Qb + (size_t)(qt0 + lhalf) * D_;
    qa0.h[0] = *reinterpret_cast<const v8h*>(qr + 0 + hb);
    qa0.h[1] = *reinterpret_cast<const v8h*>(qr + 16 + hb);
    qa1.h[0] = *reinterpret_cast<const v8h*>(qr + 32 + hb);
    qa1.h[1] = *reinterpret_cast<const v8h*>(qr + 48 + hb);
  }

  // bias takes only 4 values (mask is binary): precompute and select
  const float bs = (float)bscale[0];
  const float g1 = 0.5f * tanhf(rawb[h * 3 + 0]) * bs;
  const float g2 = 0.5f * tanhf(rawb[h * 3 + 1]) * bs;
  const float g3 = 0.5f * tanhf(rawb[h * 3 + 2]) * bs;
  const float cg1   = clamp_bias(g1);
  const float cg2   = clamp_bias(g2);
  const float cg123 = clamp_bias(g1 + g2 + g3);

  float bi0[8], bi1[8];   // per accumulator row: bias for mj=0 / mj=1
#pragma unroll
  for (int r = 0; r < 8; ++r) {
    const bool mi = maskp[b * T_ + qt0 + mrb + r] != 0;
    bi0[r] = mi ? cg1 : 0.0f;
    bi1[r] = mi ? cg123 : cg2;
  }

  float lsum[8];
#pragma unroll
  for (int r = 0; r < 8; ++r) lsum[r] = 0.0f;
  v8f o0 = {}, o1 = {}, o2 = {}, o3 = {};

  const int NJ = T_ / 32;

  // ---- stage tile 0 ----
  {
#pragma unroll
    for (int chunk = 0; chunk < 2; ++chunk) {
      const int e = (tid + chunk * 128) * 8;
      { const int r = e >> 6, c = e & 63;
        async_ld_b128(lds_off(&kts[0][r][c]), (const void*)(Kb + (size_t)r * D_ + c)); }
      { const int d = e >> 5, c = e & 31;
        async_ld_b128(lds_off(&vtt[0][d][c]), (const void*)(Vtb + (size_t)d * T_ + c)); }
    }
    wait_async0();
    __syncthreads();
  }

  for (int jb = 0; jb < NJ; ++jb) {
    const int cur = jb & 1;
    const int nxt = cur ^ 1;

    // ---- prefetch next tile while computing on current ----
    if (jb + 1 < NJ) {
      const int kb0 = (jb + 1) * 32;
#pragma unroll
      for (int chunk = 0; chunk < 2; ++chunk) {
        const int e = (tid + chunk * 128) * 8;
        { const int r = e >> 6, c = e & 63;
          async_ld_b128(lds_off(&kts[nxt][r][c]), (const void*)(Kb + (size_t)(kb0 + r) * D_ + c)); }
        { const int d = e >> 5, c = e & 31;
          async_ld_b128(lds_off(&vtt[nxt][d][c]), (const void*)(Vtb + (size_t)d * T_ + kb0 + c)); }
      }
    }

    // ---- S = Q K^T for keys [jb*32, jb*32+32) ----
    v8f s0 = {}, s1 = {};
    {
      Frag bk;
      bk.h[0] = *reinterpret_cast<const v8h*>(&kts[cur][lhalf][0 + hb]);
      bk.h[1] = *reinterpret_cast<const v8h*>(&kts[cur][lhalf][16 + hb]);
      s0 = __builtin_amdgcn_wmma_f32_16x16x32_f16(false, qa0.v, false, bk.v, (short)0, s0, false, false);
      bk.h[0] = *reinterpret_cast<const v8h*>(&kts[cur][lhalf][32 + hb]);
      bk.h[1] = *reinterpret_cast<const v8h*>(&kts[cur][lhalf][48 + hb]);
      s0 = __builtin_amdgcn_wmma_f32_16x16x32_f16(false, qa1.v, false, bk.v, (short)0, s0, false, false);
      bk.h[0] = *reinterpret_cast<const v8h*>(&kts[cur][16 + lhalf][0 + hb]);
      bk.h[1] = *reinterpret_cast<const v8h*>(&kts[cur][16 + lhalf][16 + hb]);
      s1 = __builtin_amdgcn_wmma_f32_16x16x32_f16(false, qa0.v, false, bk.v, (short)0, s1, false, false);
      bk.h[0] = *reinterpret_cast<const v8h*>(&kts[cur][16 + lhalf][32 + hb]);
      bk.h[1] = *reinterpret_cast<const v8h*>(&kts[cur][16 + lhalf][48 + hb]);
      s1 = __builtin_amdgcn_wmma_f32_16x16x32_f16(false, qa1.v, false, bk.v, (short)0, s1, false, false);
    }

    const bool mj0 = maskp[b * T_ + jb * 32 + lhalf] != 0;
    const bool mj1 = maskp[b * T_ + jb * 32 + 16 + lhalf] != 0;

    // ---- fixed-max softmax numerators, write P tile ----
#pragma unroll
    for (int r = 0; r < 8; ++r) {
      const float p0 = __expf(s0[r] + (mj0 ? bi1[r] : bi0[r]));
      const float p1 = __expf(s1[r] + (mj1 ? bi1[r] : bi0[r]));
      lsum[r] += p0 + p1;          // per-lane partial; reduced once at the end
      pts[wave][mrb + r][lhalf]      = (_Float16)p0;
      pts[wave][mrb + r][16 + lhalf] = (_Float16)p1;
    }
#if __has_builtin(__builtin_amdgcn_wave_barrier)
    __builtin_amdgcn_wave_barrier();  // keep P stores before P loads (intra-wave, LDS in-order)
#endif

    // ---- O += P V : P is A (16x32), V^T rows give contiguous B fragments ----
    Frag pa;
    pa.h[0] = *reinterpret_cast<const v8h*>(&pts[wave][lhalf][hb]);
    pa.h[1] = *reinterpret_cast<const v8h*>(&pts[wave][lhalf][16 + hb]);
#pragma unroll
    for (int dc = 0; dc < 4; ++dc) {
      Frag vb;
      vb.h[0] = *reinterpret_cast<const v8h*>(&vtt[cur][dc * 16 + lhalf][hb]);
      vb.h[1] = *reinterpret_cast<const v8h*>(&vtt[cur][dc * 16 + lhalf][16 + hb]);
      v8f* op = (dc == 0) ? &o0 : (dc == 1) ? &o1 : (dc == 2) ? &o2 : &o3;
      *op = __builtin_amdgcn_wmma_f32_16x16x32_f16(false, pa.v, false, vb.v, (short)0, *op, false, false);
    }

    // ---- retire prefetch, publish next tile ----
    wait_async0();
    __syncthreads();
  }

  // ---- epilogue: single row-sum reduction, normalize, store f16 [B,T,C] ----
#pragma unroll
  for (int r = 0; r < 8; ++r) {
    float ls = lsum[r];
    ls += __shfl_xor(ls, 1, 32);
    ls += __shfl_xor(ls, 2, 32);
    ls += __shfl_xor(ls, 4, 32);
    ls += __shfl_xor(ls, 8, 32);
    const float inv = 1.0f / ls;
    const int t = qt0 + mrb + r;
    _Float16* yrow = yh + ((size_t)(b * T_ + t)) * C_ + h * D_;
    yrow[0  + lhalf] = (_Float16)(o0[r] * inv);
    yrow[16 + lhalf] = (_Float16)(o1[r] * inv);
    yrow[32 + lhalf] = (_Float16)(o2[r] * inv);
    yrow[48 + lhalf] = (_Float16)(o3[r] * inv);
  }
}

// ---------------------------------------------------------------------------
// Output projection: out = Y @ Wproj^T, f32 result.
// grid: ((B*T)/16, C/64); block: 128 (4 waves x 16x16 tiles).
// ---------------------------------------------------------------------------
__global__ __launch_bounds__(128)
void proj_kernel(const _Float16* __restrict__ yhp,
                 const _Float16* __restrict__ wph,
                 float* __restrict__ out) {
  __shared__ __align__(16) _Float16 ys[16][392];

  const int tid  = threadIdx.x;
  const int bt0  = blockIdx.x * 16;
  const int nb   = blockIdx.y * 64;

  const _Float16* yrow = yhp + (size_t)bt0 * C_;
  for (int i = tid; i < 768; i += 128) {
    int r = i / 48, cc = (i % 48) * 8;
    async_ld_b128(lds_off(&ys[r][cc]), (const void*)(yrow + (size_t)r * C_ + cc));
  }
  wait_async0();
  __syncthreads();

  const int wave  = tid >> 5;
  const int lane  = tid & 31;
  const int lhalf = lane & 15;
  const int hb    = (lane >> 4) * 8;
  const int mrb   = (lane >> 4) * 8;
  const int n0    = nb + wave * 16;

  const _Float16* wrow = wph + (size_t)(n0 + lhalf) * C_;

  v8f acc = {};
#pragma unroll
  for (int kk = 0; kk < 12; ++kk) {
    const int kb = kk * 32;
    Frag a, bf;
    a.h[0]  = *reinterpret_cast<const v8h*>(&ys[lhalf][kb + hb]);
    a.h[1]  = *reinterpret_cast<const v8h*>(&ys[lhalf][kb + 16 + hb]);
    bf.h[0] = *reinterpret_cast<const v8h*>(wrow + kb + hb);
    bf.h[1] = *reinterpret_cast<const v8h*>(wrow + kb + 16 + hb);
    acc = __builtin_amdgcn_wmma_f32_16x16x32_f16(false, a.v, false, bf.v,
                                                 (short)0, acc, false, false);
  }

#pragma unroll
  for (int r = 0; r < 8; ++r)
    out[(size_t)(bt0 + mrb + r) * C_ + n0 + lhalf] = acc[r];
}

// ---------------------------------------------------------------------------
extern "C" void kernel_launch(void* const* d_in, const int* in_sizes, int n_in,
                              void* d_out, int out_size, void* d_ws, size_t ws_size,
                              hipStream_t stream) {
  (void)in_sizes; (void)n_in; (void)out_size; (void)ws_size;

  const float* x      = (const float*)d_in[0];
  const float* cosp   = (const float*)d_in[1];
  const float* sinp   = (const float*)d_in[2];
  const int*   maskp  = (const int*)d_in[3];
  const float* Wq     = (const float*)d_in[4];
  const float* Wk     = (const float*)d_in[5];
  const float* Wv     = (const float*)d_in[6];
  const float* Wproj  = (const float*)d_in[7];
  const float* rawb   = (const float*)d_in[8];
  const int*   bscale = (const int*)d_in[9];
  float* out = (float*)d_out;

  const size_t XN = (size_t)B_ * T_ * C_;   // 3,145,728
  const size_t WN = (size_t)C_ * C_;        // 147,456

  _Float16* p   = (_Float16*)d_ws;
  _Float16* xh  = p; p += XN;
  _Float16* wqh = p; p += WN;
  _Float16* wkh = p; p += WN;
  _Float16* wvh = p; p += WN;
  _Float16* wph = p; p += WN;
  _Float16* qh  = p; p += XN;
  _Float16* kh  = p; p += XN;
  _Float16* vh  = p; p += XN;   // transposed layout [B,H,D,T]
  _Float16* yh  = p; p += XN;

  cvt_kernel<<<2048, 256, 0, stream>>>(x, xh, (int)XN);
  cvt_kernel<<<288, 256, 0, stream>>>(Wq, wqh, (int)WN);
  cvt_kernel<<<288, 256, 0, stream>>>(Wk, wkh, (int)WN);
  cvt_kernel<<<288, 256, 0, stream>>>(Wv, wvh, (int)WN);
  cvt_kernel<<<288, 256, 0, stream>>>(Wproj, wph, (int)WN);

  dim3 g1(T_ / 16, 3 * B_);
  qkv_kernel<<<g1, 128, 0, stream>>>(xh, wqh, wkh, wvh, cosp, sinp, qh, kh, vh);

  dim3 g2(T_ / 64, H_, B_);
  attn_kernel<<<g2, 128, 0, stream>>>(qh, kh, vh, maskp, rawb, bscale, yh);

  dim3 g3((B_ * T_) / 16, C_ / 64);
  proj_kernel<<<g3, 128, 0, stream>>>(yh, wph, out);
}
